// LSTMClassifier_32633161515675
// MI455X (gfx1250) — matrix operand order repaired and checked by reading, not compiled, not run on hardware
//
#include <hip/hip_runtime.h>

typedef __attribute__((ext_vector_type(16))) _Float16 v16h;
typedef __attribute__((ext_vector_type(8)))  _Float16 v8h;
typedef __attribute__((ext_vector_type(4)))  _Float16 v4h;
typedef __attribute__((ext_vector_type(8)))  float    v8f;
typedef __attribute__((ext_vector_type(4)))  float    v4f;

#define B_TOT   64
#define SEQ     2048
#define EMB     128
#define HID     128
#define KTOT    256      // EMB + HID
#define BPW     16       // batch rows per workgroup (one 16-row M tile)
#define NWG     (B_TOT / BPW)
#define THREADS 512      // 16 wave32 waves: one 16x16 C tile each (2 gates x 8 N-tiles)

union FragU { v16h v; v8h h[2]; };

// A fragment: 16x32 f16, row-major source with leading dim `ld` (halves).
__device__ inline v16h load_A(const _Float16* src, int ld, int k0, int lane) {
  const int m = lane & 15, hi = lane >> 4;
  const _Float16* p = src + m * ld + k0 + hi * 8;
  FragU u;
  u.h[0] = *(const v8h*)p;
  u.h[1] = *(const v8h*)(p + 16);
  return u.v;
}

// Native CDNA5 V_TANH_F32 (TRANS op) — confirmed lowering to v_tanh_f32_e32.
__device__ inline float fast_tanh(float v) {
#if __has_builtin(__builtin_amdgcn_tanhf)
  return __builtin_amdgcn_tanhf(v);
#else
  float r;
  asm volatile("v_tanh_f32 %0, %1\n\tv_nop\n\tv_nop" : "=v"(r) : "v"(v));
  return r;
#endif
}

// sigmoid via native v_exp_f32 (exp2) + v_rcp_f32
__device__ inline float fast_sigmoid(float v) {
  return __builtin_amdgcn_rcpf(1.f + __builtin_amdgcn_exp2f(-1.4426950408889634f * v));
}

__global__ __launch_bounds__(THREADS)
void lstm_persistent(const int* __restrict__ x, const float* __restrict__ emb,
                     const float* __restrict__ Wf, const float* __restrict__ bf,
                     const float* __restrict__ Wc, const float* __restrict__ bc,
                     const float* __restrict__ Wfc, const float* __restrict__ bfc,
                     float* __restrict__ out)
{
  __shared__ __align__(16) _Float16 sH[BPW * HID];     // 4 KB, h_t (f16)
  __shared__ __align__(16) _Float16 sX[2][BPW * EMB];  // 8 KB, embed double buffer
  __shared__ __align__(16) float    sG[BPW * HID];     // 8 KB gate pre-act staging
  __shared__ __align__(16) float    sC[BPW * HID];     // 8 KB cand pre-act staging
  __shared__ __align__(16) float    sbf[HID];
  __shared__ __align__(16) float    sbc[HID];

  const int tid  = threadIdx.x;
  const int lane = tid & 31;
  const int wid  = tid >> 5;           // 0..15
  const int B0   = blockIdx.x * BPW;   // batch base for this WGP

  const int n0 = (wid & 7) * 16;
  const int nl = lane & 15, hi = lane >> 4;

  // ---- B (weight) fragments -> REGISTERS, once. W row-major [N][KTOT] is
  // exactly B^T, so lane (n = n0+nl) holds K = kc*32 + hi*8 .. +7 and kc*32 + 16 + hi*8 .. +7.
  const float* Wsrc = (wid & 8) ? Wc : Wf;   // waves 0-7: gate, 8-15: candidate
  v16h Breg[8];
  #pragma unroll
  for (int kc = 0; kc < 8; ++kc) {
    const float* p = Wsrc + (n0 + nl) * KTOT + kc * 32 + hi * 8;
    v4f w0 = *(const v4f*)(p + 0);
    v4f w1 = *(const v4f*)(p + 4);
    v4f w2 = *(const v4f*)(p + 16);
    v4f w3 = *(const v4f*)(p + 20);
    FragU u;
    #pragma unroll
    for (int j = 0; j < 4; ++j) {
      u.h[0][j]     = (_Float16)w0[j];
      u.h[0][4 + j] = (_Float16)w1[j];
      u.h[1][j]     = (_Float16)w2[j];
      u.h[1][4 + j] = (_Float16)w3[j];
    }
    Breg[kc] = u.v;
  }

  if (tid < HID) { sbf[tid] = bf[tid]; sbc[tid] = bc[tid]; }
  { v4h z = {}; *(v4h*)&sH[tid * 4] = z; }          // h0 = 0 (512*4 == 2048)

  // fixed elementwise / prefetch ownership: row br, 4 contiguous cols at c4
  const int br = tid >> 5;                // 0..15
  const int c4 = lane * 4;                // 0..124
  const int rowbase = (B0 + br) * SEQ;    // token row base for this thread
  const int idx0 = br * HID + c4;

  { // embed t = 0 straight into buffer 0
    const int tok = x[rowbase + 0];
    v4f e = *(const v4f*)(emb + (size_t)tok * EMB + c4);
    v4h o;
    #pragma unroll
    for (int j = 0; j < 4; ++j) o[j] = (_Float16)e[j];
    *(v4h*)&sX[0][br * EMB + c4] = o;
  }
  // 2-deep register pipeline: eReg = embed data for t+1, tokP = token for t+2
  v4f eReg;
  int tokP;
  {
    const int tok1 = x[rowbase + 1];
    eReg = *(const v4f*)(emb + (size_t)tok1 * EMB + c4);
    tokP = x[rowbase + 2];
  }
  float creg[4] = {0.f, 0.f, 0.f, 0.f};             // cell state lives in registers
  const v8f vzero = {};
  __syncthreads();

  float* stage = (wid & 8) ? sC : sG;

  // x-projection accumulator for step 0 (independent of h)
  v8f accX = vzero;
  #pragma unroll
  for (int kc = 0; kc < 4; ++kc) {
    v16h Af = load_A(&sX[0][0], EMB, kc * 32, lane);
    accX = __builtin_amdgcn_wmma_f32_16x16x32_f16(
               false, Af, false, Breg[kc], (short)0, accX, false, false);
  }

  for (int t = 0; t < SEQ; ++t) {
    const int cur = t & 1, nxt = cur ^ 1;
    (void)cur;

    // (a) commit embed data for t+1 (global load issued one iteration ago)
    {
      v4h o;
      #pragma unroll
      for (int j = 0; j < 4; ++j) o[j] = (_Float16)eReg[j];
      *(v4h*)&sX[nxt][br * EMB + c4] = o;
    }

    // (b) h-part: only 4 WMMAs sit on the recurrent critical path
    v8f acc = accX;
    #pragma unroll
    for (int kc = 4; kc < 8; ++kc) {
      v16h Af = load_A(sH, HID, kc * 32 - EMB, lane);
      acc = __builtin_amdgcn_wmma_f32_16x16x32_f16(
                false, Af, false, Breg[kc], (short)0, acc, false, false);
    }

    // (c) issue next-stage global loads: embed for t+2 (token resident), token t+3
    {
      const int t3 = (t + 3 < SEQ) ? (t + 3) : (SEQ - 1);
      eReg = *(const v4f*)(emb + (size_t)tokP * EMB + c4);
      tokP = x[rowbase + t3];
    }

    #pragma unroll
    for (int r = 0; r < 8; ++r)                  // C layout: m = r + 8*hi, n = nl
      stage[(r + 8 * hi) * HID + n0 + nl] = acc[r];
    __syncthreads();

    // (d) x-part WMMAs for step t+1 — overlaps the elementwise phase,
    //     completely off the h-dependency chain
    accX = vzero;
    #pragma unroll
    for (int kc = 0; kc < 4; ++kc) {
      v16h Af = load_A(&sX[nxt][0], EMB, kc * 32, lane);
      accX = __builtin_amdgcn_wmma_f32_16x16x32_f16(
                 false, Af, false, Breg[kc], (short)0, accX, false, false);
    }

    // (e) elementwise LSTM update (c in registers), vector LDS traffic
    {
      v4f gp4 = *(v4f*)&sG[idx0];
      v4f cp4 = *(v4f*)&sC[idx0];
      v4f bf4 = *(const v4f*)&sbf[c4];
      v4f bc4 = *(const v4f*)&sbc[c4];
      v4h h4;
      #pragma unroll
      for (int j = 0; j < 4; ++j) {
        const float g  = fast_sigmoid(gp4[j] + bf4[j]);     // shared gate (source bug kept)
        const float ct = fast_tanh(cp4[j] + bc4[j]);
        creg[j] = g * (creg[j] + ct);                       // c_new = g*(c + c_tilde)
        h4[j] = (_Float16)(g * fast_tanh(creg[j]));
      }
      *(v4h*)&sH[idx0] = h4;
    }
    __syncthreads();
  }

  // ---- final classifier: sigmoid(h_T @ Wfc.T + bfc), OUT=1 ----
  if (tid < BPW) {
    float s = bfc[0];
    #pragma unroll 8
    for (int n = 0; n < HID; ++n)
      s += (float)sH[tid * HID + n] * Wfc[n];
    out[B0 + tid] = fast_sigmoid(s);
  }
}

extern "C" void kernel_launch(void* const* d_in, const int* in_sizes, int n_in,
                              void* d_out, int out_size, void* d_ws, size_t ws_size,
                              hipStream_t stream) {
  const int*   x   = (const int*)d_in[0];
  const float* emb = (const float*)d_in[1];
  const float* Wf  = (const float*)d_in[2];
  const float* bf  = (const float*)d_in[3];
  const float* Wc  = (const float*)d_in[4];
  const float* bc  = (const float*)d_in[5];
  const float* Wfc = (const float*)d_in[6];
  const float* bfc = (const float*)d_in[7];
  float* out = (float*)d_out;
  (void)in_sizes; (void)n_in; (void)out_size; (void)d_ws; (void)ws_size;

  lstm_persistent<<<NWG, THREADS, 0, stream>>>(x, emb, Wf, bf, Wc, bc, Wfc, bfc, out);
}
